// GCNNet_33904471834793
// MI455X (gfx1250) — matrix operand order repaired and checked by reading, not compiled, not run on hardware
//
#include <hip/hip_runtime.h>
#include <hip/hip_bf16.h>
#include <cstdint>
#include <cstddef>

typedef __attribute__((ext_vector_type(16))) __bf16 v16bf;
typedef __attribute__((ext_vector_type(8)))  __bf16 v8bf;
typedef __attribute__((ext_vector_type(8)))  float  v8f;

// ---------------------------------------------------------------------------
// Split-bf16 GEMM with fragment-major LDS staging.
// Block = 128 threads (4 waves). Block tile: 64 (M) x 32 (N); each wave owns a
// 16x32 strip (two 16x16 WMMA accumulators). K-step = 32.
// LDS holds tiles already permuted into the CDNA5 wave32 WMMA fragment layout,
// so each lane's fragment load is one contiguous 32-byte read.
//
// A fragment map (16-bit A 16x32): lane<16: e=0..7 -> k=0..7,  e=8..15 -> k=16..23
//                                  lane>=16: e=0..7 -> k=8..15, e=8..15 -> k=24..31
// B fragment map (16-bit B 32x16): lane<16: e=0..7 -> k=0..7,  e=8..15 -> k=16..23 (col = lane)
//                                  lane>=16: e=0..7 -> k=8..15, e=8..15 -> k=24..31 (col = lane-16)
// C/D map: VGPR r -> row r (lanes 0-15) / row 8+r (lanes 16-31); col = lane&15
// ---------------------------------------------------------------------------
#define GEMM_TM 64
#define GEMM_TN 32
#define GEMM_TK 32

static __device__ __forceinline__ void bf16split8(const float* __restrict__ v,
                                                  v8bf& hi, v8bf& lo) {
    #pragma unroll
    for (int j = 0; j < 8; ++j) {
        __bf16 h = (__bf16)v[j];
        hi[j] = h;
        lo[j] = (__bf16)(v[j] - (float)h);
    }
}

// chunk id (0..255) -> fragment-major A slot: wave, lane, e0
static __device__ __forceinline__ void a_slot(int c, int& wv, int& lane, int& e0,
                                              int& r, int& kk0) {
    r = c >> 2;           // 0..63 tile row
    int q = c & 3;        // which 8-wide k group
    kk0 = q * 8;
    wv = r >> 4;
    int m = r & 15;
    lane = m + ((q & 1) ? 16 : 0);
    e0 = (q & 2) ? 8 : 0;
}

// chunk id (0..127) -> fragment-major B slot: tile t, lane, e0
static __device__ __forceinline__ void b_slot(int c, int& t, int& lane, int& e0,
                                              int& nn, int& kk0) {
    nn = c & 31;          // 0..31 tile col
    int kg = c >> 5;      // 0..3
    kk0 = kg * 8;
    t = nn >> 4;
    int n = nn & 15;
    lane = n + ((kg & 1) ? 16 : 0);
    e0 = (kg & 2) ? 8 : 0;
}

template<bool RELU_IN, bool RELU_OUT>
__global__ __launch_bounds__(128)
void gemm_bf16s(const float* __restrict__ A, const float* __restrict__ W,
                const float* __restrict__ bias, float* __restrict__ C,
                int M, int N, int K, int ldc) {
    __shared__ __bf16 Ah[4][32][16];   // [wave][lane][elem]
    __shared__ __bf16 Al[4][32][16];
    __shared__ __bf16 Bh[2][32][16];   // [ntile][lane][elem]
    __shared__ __bf16 Bl[2][32][16];

    const int tid  = threadIdx.x;
    const int lane = tid & 31;
    const int wv   = tid >> 5;
    const int m0   = blockIdx.x * GEMM_TM;
    const int n0   = blockIdx.y * GEMM_TN;
    const bool doN1 = (n0 + 16) < N;   // block-uniform: second 16-wide n-tile live?

    v8f acc0 = {}, acc1 = {};

    for (int k0 = 0; k0 < K; k0 += GEMM_TK) {
        // ---- stage A tile (64x32) as 8-element chunks, 2 chunks/thread ----
        #pragma unroll
        for (int cc = 0; cc < 2; ++cc) {
            int c = tid + cc * 128;
            int awv, aln, ae0, r, kk0;
            a_slot(c, awv, aln, ae0, r, kk0);
            int gr = m0 + r, gk0 = k0 + kk0;
            float v[8];
            #pragma unroll
            for (int j = 0; j < 8; ++j) {
                float t = 0.0f;
                if (gr < M && (gk0 + j) < K) t = A[(size_t)gr * K + gk0 + j];
                if (RELU_IN) t = fmaxf(t, 0.0f);
                v[j] = t;
            }
            v8bf hi, lo;
            bf16split8(v, hi, lo);
            *(v8bf*)&Ah[awv][aln][ae0] = hi;
            *(v8bf*)&Al[awv][aln][ae0] = lo;
        }
        // ---- stage B tile (32x32), 1 chunk/thread ----
        {
            int bt, bln, be0, nn, kk0;
            b_slot(tid, bt, bln, be0, nn, kk0);
            int gn = n0 + nn, gk0 = k0 + kk0;
            float v[8];
            #pragma unroll
            for (int j = 0; j < 8; ++j) {
                float t = 0.0f;
                if (gn < N && (gk0 + j) < K) t = W[(size_t)(gk0 + j) * N + gn];
                v[j] = t;
            }
            v8bf hi, lo;
            bf16split8(v, hi, lo);
            *(v8bf*)&Bh[bt][bln][be0] = hi;
            *(v8bf*)&Bl[bt][bln][be0] = lo;
        }
        __syncthreads();

        // ---- contiguous 32B fragment loads ----
        v16bf ah  = *(const v16bf*)&Ah[wv][lane][0];
        v16bf al  = *(const v16bf*)&Al[wv][lane][0];
        v16bf bh0 = *(const v16bf*)&Bh[0][lane][0];
        v16bf bl0 = *(const v16bf*)&Bl[0][lane][0];
        v16bf bh1 = *(const v16bf*)&Bh[1][lane][0];
        v16bf bl1 = *(const v16bf*)&Bl[1][lane][0];
        __syncthreads();

        acc0 = __builtin_amdgcn_wmma_f32_16x16x32_bf16(false, ah, false, bh0, (short)0, acc0, false, false);
        acc0 = __builtin_amdgcn_wmma_f32_16x16x32_bf16(false, al, false, bh0, (short)0, acc0, false, false);
        acc0 = __builtin_amdgcn_wmma_f32_16x16x32_bf16(false, ah, false, bl0, (short)0, acc0, false, false);
        if (doN1) {
            acc1 = __builtin_amdgcn_wmma_f32_16x16x32_bf16(false, ah, false, bh1, (short)0, acc1, false, false);
            acc1 = __builtin_amdgcn_wmma_f32_16x16x32_bf16(false, al, false, bh1, (short)0, acc1, false, false);
            acc1 = __builtin_amdgcn_wmma_f32_16x16x32_bf16(false, ah, false, bl1, (short)0, acc1, false, false);
        }
    }

    // ---- epilogue ----
    const int rbase = (lane < 16) ? 0 : 8;
    #pragma unroll
    for (int t = 0; t < 2; ++t) {
        int col = n0 + t * 16 + (lane & 15);
        if (col < N) {
            const v8f& acc = t ? acc1 : acc0;
            float bv = bias ? bias[col] : 0.0f;
            #pragma unroll
            for (int r = 0; r < 8; ++r) {
                int row = m0 + wv * 16 + rbase + r;
                if (row < M) {
                    float v = acc[r] + bv;
                    if (RELU_OUT) v = fmaxf(v, 0.0f);
                    C[(size_t)row * ldc + col] = v;
                }
            }
        }
    }
}

// ---------------------------------------------------------------------------
// Conv branch as implicit-im2col GEMM:
//   c[(b,w), o] = sum_{i,k} emb[target[b,i], w+k] * convw[o, i, k]  (+ convb)
// M = B*121, K = 8000 (i*8+k), N = 32. A k-chunk of 8 is exactly one tap row
// emb[tok, w..w+7] (contiguous). Output stored as xtA[b, o*121 + w].
// ---------------------------------------------------------------------------
__global__ __launch_bounds__(128)
void conv_im2col_gemm(const float* __restrict__ emb, const int* __restrict__ target,
                      const float* __restrict__ convw, const float* __restrict__ convb,
                      float* __restrict__ xtA, int Bg) {
    const int K = 8000;
    const int M = Bg * 121;

    __shared__ __bf16 Ah[4][32][16];
    __shared__ __bf16 Al[4][32][16];
    __shared__ __bf16 Bh[2][32][16];
    __shared__ __bf16 Bl[2][32][16];

    const int tid  = threadIdx.x;
    const int lane = tid & 31;
    const int wv   = tid >> 5;
    const int m0   = blockIdx.x * GEMM_TM;

    v8f acc0 = {}, acc1 = {};

    for (int k0 = 0; k0 < K; k0 += GEMM_TK) {
        #pragma unroll
        for (int cc = 0; cc < 2; ++cc) {
            int c = tid + cc * 128;
            int awv, aln, ae0, r, kk0;
            a_slot(c, awv, aln, ae0, r, kk0);
            int gr = m0 + r;
            float v[8];
            if (gr < M) {
                int b = gr / 121, w = gr - b * 121;
                int i = (k0 + kk0) >> 3;                     // chunk == one tap row
                int tok = target[(size_t)b * 1000 + i];      // in [0,26)
                const float* base = emb + (size_t)tok * 128 + w;  // w+7 <= 127
                #pragma unroll
                for (int j = 0; j < 8; ++j) v[j] = base[j];
            } else {
                #pragma unroll
                for (int j = 0; j < 8; ++j) v[j] = 0.0f;
            }
            v8bf hi, lo;
            bf16split8(v, hi, lo);
            *(v8bf*)&Ah[awv][aln][ae0] = hi;
            *(v8bf*)&Al[awv][aln][ae0] = lo;
        }
        {
            int bt, bln, be0, nn, kk0;
            b_slot(tid, bt, bln, be0, nn, kk0);
            int gk0 = k0 + kk0;
            const float* wp = convw + (size_t)nn * 8000 + gk0;  // (O,I,W) row-major
            float v[8];
            #pragma unroll
            for (int j = 0; j < 8; ++j) v[j] = wp[j];
            v8bf hi, lo;
            bf16split8(v, hi, lo);
            *(v8bf*)&Bh[bt][bln][be0] = hi;
            *(v8bf*)&Bl[bt][bln][be0] = lo;
        }
        __syncthreads();

        v16bf ah  = *(const v16bf*)&Ah[wv][lane][0];
        v16bf al  = *(const v16bf*)&Al[wv][lane][0];
        v16bf bh0 = *(const v16bf*)&Bh[0][lane][0];
        v16bf bl0 = *(const v16bf*)&Bl[0][lane][0];
        v16bf bh1 = *(const v16bf*)&Bh[1][lane][0];
        v16bf bl1 = *(const v16bf*)&Bl[1][lane][0];
        __syncthreads();

        acc0 = __builtin_amdgcn_wmma_f32_16x16x32_bf16(false, ah, false, bh0, (short)0, acc0, false, false);
        acc0 = __builtin_amdgcn_wmma_f32_16x16x32_bf16(false, al, false, bh0, (short)0, acc0, false, false);
        acc0 = __builtin_amdgcn_wmma_f32_16x16x32_bf16(false, ah, false, bl0, (short)0, acc0, false, false);
        acc1 = __builtin_amdgcn_wmma_f32_16x16x32_bf16(false, ah, false, bh1, (short)0, acc1, false, false);
        acc1 = __builtin_amdgcn_wmma_f32_16x16x32_bf16(false, al, false, bh1, (short)0, acc1, false, false);
        acc1 = __builtin_amdgcn_wmma_f32_16x16x32_bf16(false, ah, false, bl1, (short)0, acc1, false, false);
    }

    const int rbase = (lane < 16) ? 0 : 8;
    #pragma unroll
    for (int t = 0; t < 2; ++t) {
        int col = t * 16 + (lane & 15);   // 0..31
        const v8f& acc = t ? acc1 : acc0;
        float bv = convb[col];
        #pragma unroll
        for (int r = 0; r < 8; ++r) {
            int row = m0 + wv * 16 + rbase + r;
            if (row < M) {
                int b = row / 121, w = row - b * 121;
                xtA[(size_t)b * 3872 + (size_t)col * 121 + w] = acc[r] + bv;
            }
        }
    }
}

// ---------------------------------------------------------------------------
// Graph utility kernels
// ---------------------------------------------------------------------------
__global__ void fill_f32(float* __restrict__ p, float v, size_t n) {
    size_t i = (size_t)blockIdx.x * blockDim.x + threadIdx.x;
    size_t s = (size_t)gridDim.x * blockDim.x;
    for (; i < n; i += s) p[i] = v;
}

__global__ void deg_count(const int* __restrict__ dst, float* __restrict__ deg, int E) {
    int i = blockIdx.x * blockDim.x + threadIdx.x;
    int s = gridDim.x * blockDim.x;
    for (; i < E; i += s) atomicAdd(&deg[dst[i]], 1.0f);
}

__global__ void deg_to_dinv(float* __restrict__ deg, int n) {
    int i = blockIdx.x * blockDim.x + threadIdx.x;
    int s = gridDim.x * blockDim.x;
    for (; i < n; i += s) {
        float d = deg[i];
        deg[i] = (d > 0.0f) ? rsqrtf(d) : 0.0f;
    }
}

// out[i,f] = bias[f] + H[i,f] * dinv[i]^2   (self-loop term + bias init)
__global__ void scatter_self(const float* __restrict__ H, const float* __restrict__ dinv,
                             const float* __restrict__ bias, float* __restrict__ out,
                             int Nn, int F) {
    size_t total = (size_t)Nn * F;
    size_t i = (size_t)blockIdx.x * blockDim.x + threadIdx.x;
    size_t s = (size_t)gridDim.x * blockDim.x;
    for (; i < total; i += s) {
        int node = (int)(i / F);
        int f    = (int)(i - (size_t)node * F);
        float di = dinv[node];
        out[i] = bias[f] + H[i] * di * di;
    }
}

// out[dst,f] += H[src,f] * dinv[src]*dinv[dst] — one wave per edge
__global__ void scatter_edges(const int* __restrict__ src, const int* __restrict__ dst,
                              const float* __restrict__ dinv, const float* __restrict__ H,
                              float* __restrict__ out, int E, int F) {
    int warp  = (blockIdx.x * blockDim.x + threadIdx.x) >> 5;
    int lane  = threadIdx.x & 31;
    int nwarp = (gridDim.x * blockDim.x) >> 5;
    for (int e = warp; e < E; e += nwarp) {
        int s = src[e], d = dst[e];
        if (e + nwarp < E) {
            __builtin_prefetch(&H[(size_t)src[e + nwarp] * F], 0, 1);
        }
        float nrm = dinv[s] * dinv[d];
        const float* hs = H + (size_t)s * F;
        float*       od = out + (size_t)d * F;
        for (int f = lane; f < F; f += 32)
            atomicAdd(&od[f], hs[f] * nrm);
    }
}

// g[batch[i],f] = max(g, relu(H[i,f])); values >= 0 so uint-bit atomicMax is exact
__global__ void pool_max(const float* __restrict__ H, const int* __restrict__ batch,
                         unsigned int* __restrict__ g, int Nn, int F) {
    int warp  = (blockIdx.x * blockDim.x + threadIdx.x) >> 5;
    int lane  = threadIdx.x & 31;
    int nwarp = (gridDim.x * blockDim.x) >> 5;
    for (int i = warp; i < Nn; i += nwarp) {
        int b = batch[i];
        const float* hr = H + (size_t)i * F;
        unsigned int* gr = g + (size_t)b * F;
        for (int f = lane; f < F; f += 32) {
            float v = fmaxf(hr[f], 0.0f);
            atomicMax(&gr[f], __float_as_uint(v));
        }
    }
}

// ---------------------------------------------------------------------------
extern "C" void kernel_launch(void* const* d_in, const int* in_sizes, int n_in,
                              void* d_out, int out_size, void* d_ws, size_t ws_size,
                              hipStream_t stream) {
    const float* x      = (const float*)d_in[0];
    const int*   ei     = (const int*)  d_in[1];
    const int*   batch  = (const int*)  d_in[2];
    const int*   target = (const int*)  d_in[3];
    const float* W1 = (const float*)d_in[4],  *b1 = (const float*)d_in[5];
    const float* W2 = (const float*)d_in[6],  *b2 = (const float*)d_in[7];
    const float* W3 = (const float*)d_in[8],  *b3 = (const float*)d_in[9];
    const float* Wg1= (const float*)d_in[10], *bg1= (const float*)d_in[11];
    const float* Wg2= (const float*)d_in[12], *bg2= (const float*)d_in[13];
    const float* emb= (const float*)d_in[14];
    const float* convw=(const float*)d_in[15],*convb=(const float*)d_in[16];
    const float* Wxt= (const float*)d_in[17], *bxt= (const float*)d_in[18];
    const float* Wf1= (const float*)d_in[19], *bf1= (const float*)d_in[20];
    const float* Wf2= (const float*)d_in[21], *bf2= (const float*)d_in[22];
    const float* Wo = (const float*)d_in[23], *bo = (const float*)d_in[24];
    (void)n_in; (void)out_size; (void)ws_size;

    const int Nn = in_sizes[0] / 78;     // 200000
    const int E  = in_sizes[1] / 2;      // 800000
    const int Bg = in_sizes[3] / 1000;   // 512
    const int* src = ei;
    const int* dst = ei + E;

    float* ws = (float*)d_ws;
    size_t off = 0;
    auto alloc = [&](size_t n) { float* p = ws + off; off += n; return p; };
    float* dinv = alloc((size_t)Nn);
    float* bufA = alloc((size_t)Nn * 312);
    float* bufB = alloc((size_t)Nn * 312);
    float* g    = alloc((size_t)Bg * 312);
    float* gg1  = alloc((size_t)Bg * 1024);
    float* xc   = alloc((size_t)Bg * 256);
    float* xtA  = alloc((size_t)Bg * 3872);
    float* f1   = alloc((size_t)Bg * 1024);
    float* f2   = alloc((size_t)Bg * 512);

    auto gemm = [&](const float* A, const float* W, const float* bias, float* C,
                    int M, int N, int K, int ldc, bool relu_in, bool relu_out) {
        dim3 grid((M + GEMM_TM - 1) / GEMM_TM, (N + GEMM_TN - 1) / GEMM_TN);
        dim3 blk(128);
        if (relu_in) {
            if (relu_out) gemm_bf16s<true, true ><<<grid, blk, 0, stream>>>(A, W, bias, C, M, N, K, ldc);
            else          gemm_bf16s<true, false><<<grid, blk, 0, stream>>>(A, W, bias, C, M, N, K, ldc);
        } else {
            if (relu_out) gemm_bf16s<false, true ><<<grid, blk, 0, stream>>>(A, W, bias, C, M, N, K, ldc);
            else          gemm_bf16s<false, false><<<grid, blk, 0, stream>>>(A, W, bias, C, M, N, K, ldc);
        }
    };

    // --- degrees: deg = 1 (self loop) + count(dst); dinv = rsqrt(deg) ---
    fill_f32<<<1024, 256, 0, stream>>>(dinv, 1.0f, (size_t)Nn);
    deg_count<<<(E + 255) / 256, 256, 0, stream>>>(dst, dinv, E);
    deg_to_dinv<<<(Nn + 255) / 256, 256, 0, stream>>>(dinv, Nn);

    const int eblocks = (E * 32 + 255) / 256;   // one wave per edge
    const int pblocks = (Nn * 32 + 255) / 256;  // one wave per node

    // --- GCN layer 1: H = x @ W1; out = b1 + self + edges (relu deferred) ---
    gemm(x, W1, nullptr, bufA, Nn, 78, 78, 78, false, false);
    scatter_self<<<((size_t)Nn * 78 + 255) / 256, 256, 0, stream>>>(bufA, dinv, b1, bufB, Nn, 78);
    scatter_edges<<<eblocks, 256, 0, stream>>>(src, dst, dinv, bufA, bufB, E, 78);

    // --- GCN layer 2 (relu fused into GEMM A-load) ---
    gemm(bufB, W2, nullptr, bufA, Nn, 156, 78, 156, true, false);
    scatter_self<<<((size_t)Nn * 156 + 255) / 256, 256, 0, stream>>>(bufA, dinv, b2, bufB, Nn, 156);
    scatter_edges<<<eblocks, 256, 0, stream>>>(src, dst, dinv, bufA, bufB, E, 156);

    // --- GCN layer 3 ---
    gemm(bufB, W3, nullptr, bufA, Nn, 312, 156, 312, true, false);
    scatter_self<<<((size_t)Nn * 312 + 255) / 256, 256, 0, stream>>>(bufA, dinv, b3, bufB, Nn, 312);
    scatter_edges<<<eblocks, 256, 0, stream>>>(src, dst, dinv, bufA, bufB, E, 312);

    // --- segment_max pooling (relu fused; empty segments -> 0) ---
    fill_f32<<<256, 256, 0, stream>>>(g, 0.0f, (size_t)Bg * 312);
    pool_max<<<pblocks, 256, 0, stream>>>(bufB, batch, (unsigned int*)g, Nn, 312);

    // --- graph MLP: g -> 1024 (relu) -> 128, into xc[:, 0:128] ---
    gemm(g,   Wg1, bg1, gg1, Bg, 1024, 312, 1024, false, true);
    gemm(gg1, Wg2, bg2, xc,  Bg, 128, 1024, 256, false, false);

    // --- conv branch: implicit im2col GEMM, then @Wxt into xc[:, 128:256] ---
    {
        dim3 grid(((Bg * 121) + GEMM_TM - 1) / GEMM_TM, 1);
        conv_im2col_gemm<<<grid, 128, 0, stream>>>(emb, target, convw, convb, xtA, Bg);
    }
    gemm(xtA, Wxt, bxt, xc + 128, Bg, 128, 3872, 256, false, false);

    // --- head: 256 -> 1024 (relu) -> 512 (relu) -> 1 ---
    gemm(xc, Wf1, bf1, f1, Bg, 1024, 256, 1024, false, true);
    gemm(f1, Wf2, bf2, f2, Bg, 512, 1024, 512, false, true);
    gemm(f2, Wo,  bo,  (float*)d_out, Bg, 1, 512, 1, false, false);
}
